// decoder_attn_85169201480436
// MI455X (gfx1250) — compile-verified, matrix operand order checked by reading, not run on hardware
//
#include <hip/hip_runtime.h>

// MI455X / gfx1250 decoder: BF16 WMMA GEMMs (fp32 accumulate) with 4-way
// split-K per 4-wave workgroup (LDS reduction) to fix occupancy on the
// M=128-limited shapes; global_prefetch on the weight stream; fused
// attention/softmax kernels. Host loop is graph-captured by the harness.

typedef __attribute__((ext_vector_type(16))) __bf16 v16bf;
typedef __attribute__((ext_vector_type(8)))  __bf16 v8bf;
typedef __attribute__((ext_vector_type(8)))  float  v8f;

#define DEVFN static __device__ __forceinline__

DEVFN __bf16 f2bf(float x) {
  unsigned u = __builtin_bit_cast(unsigned, x);
  unsigned r = (u + 0x7FFFu + ((u >> 16) & 1u)) >> 16;
  unsigned short h = (unsigned short)r;
  return __builtin_bit_cast(__bf16, h);
}

// Load one 16x32 bf16 WMMA fragment from a row-major matrix (ld = K stride).
// Lane holds row (row0 + lane&15); elements 0..7 = K[k0+kb .. k0+kb+7],
// elements 8..15 = K[k0+kb+16 ..], kb = (lane>=16) ? 8 : 0.  (ISA 7.12.2)
DEVFN v16bf load_frag(const __bf16* __restrict__ base, int ld, int row0, int k0, int lane) {
  const __bf16* p = base + (size_t)(row0 + (lane & 15)) * ld + (k0 + ((lane >> 4) << 3));
  v8bf lo = *(const v8bf*)(p);
  v8bf hi = *(const v8bf*)(p + 16);
  v16bf a;
#pragma unroll
  for (int i = 0; i < 8; ++i) { a[i] = lo[i]; a[i + 8] = hi[i]; }
  return a;
}

// C[M,Npad] = A[M,K](bf16) * W[Npad,K]^T(bf16) + bias + addend.
// 4-wave workgroup; each wave computes the same 32x64 tile over one K-quarter
// (2 A-frags x 4 B-frags -> 8 wmma per K-step), partials reduced through LDS.
__global__ void gemm_bf16_splitk(const __bf16* __restrict__ A, const __bf16* __restrict__ W,
                                 const float* __restrict__ bias, const float* __restrict__ addend,
                                 float* __restrict__ C, int K, int Npad, int Nreal) {
  __shared__ float part[4][32][65];  // 65-float rows: conflict-free banks
  int lane = threadIdx.x & 31;
  int w    = threadIdx.x >> 5;       // K-slice id, 0..3
  int m0 = blockIdx.y * 32;
  int n0 = blockIdx.x * 64;
  int kw = K >> 2;                   // K per wave (K % 128 == 0 for all uses)
  int kbeg = w * kw, kend = kbeg + kw;
  v8f acc[2][4] = {};
  for (int kk = kbeg; kk < kend; kk += 32) {
    // prefetch weight stream two K-steps ahead (global_prefetch_b8 path)
    if (kk + 64 < kend) {
      __builtin_prefetch((const void*)(W + (size_t)(n0 + lane) * K + (kk + 64)), 0, 1);
      __builtin_prefetch((const void*)(W + (size_t)(n0 + 32 + lane) * K + (kk + 64)), 0, 1);
    }
    v16bf a0 = load_frag(A, K, m0, kk, lane);
    v16bf a1 = load_frag(A, K, m0 + 16, kk, lane);
#pragma unroll
    for (int j = 0; j < 4; ++j) {
      v16bf b = load_frag(W, K, n0 + j * 16, kk, lane);
      acc[0][j] = __builtin_amdgcn_wmma_f32_16x16x32_bf16(
          false, a0, false, b, (short)0, acc[0][j], false, false);
      acc[1][j] = __builtin_amdgcn_wmma_f32_16x16x32_bf16(
          false, a1, false, b, (short)0, acc[1][j], false, false);
    }
  }
  // spill partial tile to LDS: row = mt*16 + r + 8*(lane>>4), col = j*16 + (lane&15)
  int cl = lane & 15;
  int rh = (lane >> 4) << 3;
#pragma unroll
  for (int j = 0; j < 4; ++j)
#pragma unroll
    for (int mt = 0; mt < 2; ++mt)
#pragma unroll
      for (int r = 0; r < 8; ++r)
        part[w][mt * 16 + rh + r][j * 16 + cl] = acc[mt][j][r];
  __syncthreads();
  // 128 threads reduce 4 partials over the 32x64 tile, apply bias/addend, store
  for (int e = threadIdx.x; e < 32 * 64; e += 128) {
    int r = e >> 6, cc = e & 63;
    int col = n0 + cc;
    if (col >= Nreal) continue;
    float v = part[0][r][cc] + part[1][r][cc] + part[2][r][cc] + part[3][r][cc];
    if (bias) v += bias[col];
    size_t idx = (size_t)(m0 + r) * Npad + col;
    if (addend) v += addend[idx];
    C[idx] = v;
  }
}

__global__ void cvt_bf16(const float* __restrict__ s, __bf16* __restrict__ d, long n) {
  long i = (long)blockIdx.x * blockDim.x + threadIdx.x;
  if (i < n) d[i] = f2bf(s[i]);
}

// lin_w [7000,1024] -> bf16 padded to [7040,1024] (zeros in pad rows)
__global__ void cvt_lin(const float* __restrict__ s, __bf16* __restrict__ d) {
  long i = (long)blockIdx.x * blockDim.x + threadIdx.x;
  if (i >= (long)7040 * 1024) return;
  long r = i >> 10;
  d[i] = (r < 7000) ? f2bf(s[i]) : f2bf(0.0f);
}

// xts[l*128+b, k]: l==0 -> hidden_en[b,k]; l>=1 -> embed_w[target[b,l-1], k]
__global__ void build_xts(const float* __restrict__ hidden_en, const float* __restrict__ embed_w,
                          const int* __restrict__ target, __bf16* __restrict__ xts) {
  long i = (long)blockIdx.x * blockDim.x + threadIdx.x;
  if (i >= (long)5120 * 512) return;
  int k = (int)(i & 511);
  int row = (int)(i >> 9);
  int b = row & 127, l = row >> 7;
  float v = (l == 0) ? hidden_en[b * 512 + k]
                     : embed_w[(size_t)target[b * 40 + (l - 1)] * 512 + k];
  xts[i] = f2bf(v);
}

__global__ void init_state(float* __restrict__ h, float* __restrict__ c, __bf16* __restrict__ hbf) {
  int i = blockIdx.x * blockDim.x + threadIdx.x;
  if (i < 128 * 512) { h[i] = 0.0f; c[i] = 0.0f; hbf[i] = f2bf(0.0f); }
}

DEVFN float sigm(float x) { return 1.0f / (1.0f + expf(-x)); }

__global__ void lstm_pointwise(const float* __restrict__ gates, float* __restrict__ h,
                               float* __restrict__ c, __bf16* __restrict__ hbf) {
  int i = blockIdx.x * blockDim.x + threadIdx.x;  // 65536
  int b = i >> 9, j = i & 511;
  const float* g = gates + (size_t)b * 2048;
  float gi = g[j], gf = g[j + 512], gg = g[j + 1024], go = g[j + 1536];
  float cn = sigm(gf) * c[i] + sigm(gi) * tanhf(gg);
  float hn = sigm(go) * tanhf(cn);
  c[i] = cn; h[i] = hn; hbf[i] = f2bf(hn);
}

// Per-b fused attention: feat_att -> softmax(HW) -> context matvec; emits hcat bf16.
__global__ void attn_kernel(const float* __restrict__ x_em, const float* __restrict__ att_x_em,
                            const float* __restrict__ hem, const float* __restrict__ attw_w,
                            const float* __restrict__ attw_b, const float* __restrict__ h,
                            __bf16* __restrict__ hcat) {
  __shared__ float wv[512], hv[512], sm[256], red[256];
  int b = blockIdx.x, tid = threadIdx.x;  // 256 threads, one per hw
  wv[tid] = attw_w[tid];            wv[tid + 256] = attw_w[tid + 256];
  hv[tid] = hem[b * 512 + tid];     hv[tid + 256] = hem[b * 512 + tid + 256];
  __syncthreads();
  const float* xb = x_em + (size_t)b * 512 * 256;
  float s = 0.0f;
  for (int cc = 0; cc < 512; ++cc)
    s += wv[cc] * tanhf(xb[cc * 256 + tid] + hv[cc]);
  s += attw_b[0];
  red[tid] = s; __syncthreads();
  for (int off = 128; off > 0; off >>= 1) {
    if (tid < off) red[tid] = fmaxf(red[tid], red[tid + off]);
    __syncthreads();
  }
  float mx = red[0]; __syncthreads();
  float e = expf(s - mx);
  red[tid] = e; __syncthreads();
  for (int off = 128; off > 0; off >>= 1) {
    if (tid < off) red[tid] += red[tid + off];
    __syncthreads();
  }
  sm[tid] = e / red[0]; __syncthreads();
  const float* axb = att_x_em + (size_t)b * 256 * 512;
  float a0 = 0.0f, a1 = 0.0f;
  for (int hw = 0; hw < 256; ++hw) {
    float al = sm[hw];
    a0 += al * axb[hw * 512 + tid];
    a1 += al * axb[hw * 512 + tid + 256];
  }
  __bf16* hc = hcat + (size_t)b * 1024;
  hc[tid]       = f2bf(h[b * 512 + tid]);
  hc[tid + 256] = f2bf(h[b * 512 + tid + 256]);
  hc[512 + tid]       = f2bf(a0);
  hc[512 + tid + 256] = f2bf(a1);
}

__global__ void logsoftmax_kernel(const float* __restrict__ scores, float* __restrict__ out, int t) {
  __shared__ float red[256];
  int b = blockIdx.x, tid = threadIdx.x;
  const float* row = scores + (size_t)b * 7040;
  float m = -3.402823466e38f;
  for (int n = tid; n < 7000; n += 256) m = fmaxf(m, row[n]);
  red[tid] = m; __syncthreads();
  for (int off = 128; off > 0; off >>= 1) {
    if (tid < off) red[tid] = fmaxf(red[tid], red[tid + off]);
    __syncthreads();
  }
  float mx = red[0]; __syncthreads();
  float s = 0.0f;
  for (int n = tid; n < 7000; n += 256) s += expf(row[n] - mx);
  red[tid] = s; __syncthreads();
  for (int off = 128; off > 0; off >>= 1) {
    if (tid < off) red[tid] += red[tid + off];
    __syncthreads();
  }
  float lg = mx + logf(red[0]);
  float* orow = out + ((size_t)b * 40 + t) * 7000;
  for (int n = tid; n < 7000; n += 256) orow[n] = row[n] - lg;
}

extern "C" void kernel_launch(void* const* d_in, const int* in_sizes, int n_in,
                              void* d_out, int out_size, void* d_ws, size_t ws_size,
                              hipStream_t stream) {
  (void)in_sizes; (void)n_in; (void)out_size; (void)ws_size;
  const float* hidden_en = (const float*)d_in[0];
  const float* x_em      = (const float*)d_in[1];
  const float* att_x_em  = (const float*)d_in[2];
  const int*   target    = (const int*)d_in[3];
  const float* embed_w   = (const float*)d_in[6];
  const float* W_ih      = (const float*)d_in[7];
  const float* W_hh      = (const float*)d_in[8];
  const float* b_ih      = (const float*)d_in[9];
  const float* b_hh      = (const float*)d_in[10];
  const float* hem_w     = (const float*)d_in[11];
  const float* hem_b     = (const float*)d_in[12];
  const float* attw_w    = (const float*)d_in[13];
  const float* attw_b    = (const float*)d_in[14];
  const float* lin_w     = (const float*)d_in[15];
  const float* lin_b     = (const float*)d_in[16];
  float* out = (float*)d_out;

  char* ws = (char*)d_ws;
  size_t off = 0;
  auto alloc = [&](size_t bytes) -> void* {
    void* p = ws + off; off += (bytes + 255) & ~(size_t)255; return p;
  };
  __bf16* Wih_bf  = (__bf16*)alloc((size_t)2048 * 512 * 2);
  __bf16* Whh_bf  = (__bf16*)alloc((size_t)2048 * 512 * 2);
  __bf16* hemw_bf = (__bf16*)alloc((size_t)512 * 512 * 2);
  __bf16* lin_bf  = (__bf16*)alloc((size_t)7040 * 1024 * 2);
  __bf16* xts_bf  = (__bf16*)alloc((size_t)5120 * 512 * 2);
  float*  xg_all  = (float*)alloc((size_t)5120 * 2048 * 4);
  float*  gates   = (float*)alloc((size_t)128 * 2048 * 4);
  float*  h       = (float*)alloc((size_t)128 * 512 * 4);
  float*  c       = (float*)alloc((size_t)128 * 512 * 4);
  __bf16* h_bf    = (__bf16*)alloc((size_t)128 * 512 * 2);
  float*  hem     = (float*)alloc((size_t)128 * 512 * 4);
  __bf16* hcat_bf = (__bf16*)alloc((size_t)128 * 1024 * 2);
  float*  scores  = (float*)alloc((size_t)128 * 7040 * 4);

  // ---- prep ----
  cvt_bf16<<<(2048 * 512 + 255) / 256, 256, 0, stream>>>(W_ih, Wih_bf, (long)2048 * 512);
  cvt_bf16<<<(2048 * 512 + 255) / 256, 256, 0, stream>>>(W_hh, Whh_bf, (long)2048 * 512);
  cvt_bf16<<<(512 * 512 + 255) / 256, 256, 0, stream>>>(hem_w, hemw_bf, (long)512 * 512);
  cvt_lin<<<(7040 * 1024 + 255) / 256, 256, 0, stream>>>(lin_w, lin_bf);
  build_xts<<<(5120 * 512 + 255) / 256, 256, 0, stream>>>(hidden_en, embed_w, target, xts_bf);
  init_state<<<(128 * 512) / 256, 256, 0, stream>>>(h, c, h_bf);
  // xg_all[L*B, 2048] = xts @ W_ih^T + b_ih  (one big WMMA GEMM, M=5120)
  gemm_bf16_splitk<<<dim3(2048 / 64, 5120 / 32), 128, 0, stream>>>(
      xts_bf, Wih_bf, b_ih, nullptr, xg_all, 512, 2048, 2048);

  // ---- 40 recurrent steps ----
  for (int t = 0; t < 40; ++t) {
    // gates = xg_all[t] + h @ W_hh^T + b_hh
    gemm_bf16_splitk<<<dim3(2048 / 64, 128 / 32), 128, 0, stream>>>(
        h_bf, Whh_bf, b_hh, xg_all + (size_t)t * 128 * 2048, gates, 512, 2048, 2048);
    lstm_pointwise<<<(128 * 512) / 256, 256, 0, stream>>>(gates, h, c, h_bf);
    // hem = h @ hem_w^T + hem_b
    gemm_bf16_splitk<<<dim3(512 / 64, 128 / 32), 128, 0, stream>>>(
        h_bf, hemw_bf, hem_b, nullptr, hem, 512, 512, 512);
    attn_kernel<<<128, 256, 0, stream>>>(x_em, att_x_em, hem, attw_w, attw_b, h, hcat_bf);
    // scores = [h, att] @ lin_w^T + lin_b   (N padded to 7040)
    gemm_bf16_splitk<<<dim3(7040 / 64, 128 / 32), 128, 0, stream>>>(
        hcat_bf, lin_bf, lin_b, nullptr, scores, 1024, 7040, 7000);
    logsoftmax_kernel<<<128, 256, 0, stream>>>(scores, out, t);
  }
}
